// Transformer_58995670778146
// MI455X (gfx1250) — matrix operand order repaired and checked by reading, not compiled, not run on hardware
//
#include <hip/hip_runtime.h>
#include <math.h>

// ---------- types ----------
typedef __bf16 bf16;
typedef __attribute__((ext_vector_type(16))) __bf16 v16bf;
typedef __attribute__((ext_vector_type(8)))  __bf16 v8bf;
typedef __attribute__((ext_vector_type(8)))  float  v8f;
typedef __attribute__((ext_vector_type(4)))  unsigned int u32x4;
typedef __attribute__((ext_vector_type(8)))  int i32x8;
typedef __attribute__((ext_vector_type(4)))  int i32x4;

#define CAT16(lo,hi) __builtin_shufflevector(lo,hi,0,1,2,3,4,5,6,7,8,9,10,11,12,13,14,15)

#if defined(__has_builtin)
# if __has_builtin(__builtin_amdgcn_tensor_load_to_lds)
#  define HAVE_TDM 1
# endif
#endif
#ifndef HAVE_TDM
# define HAVE_TDM 0
#endif

// ---------- problem sizes ----------
constexpr int D    = 256;
constexpr int Hh   = 8;
constexpr int NL   = 2;
constexpr int BATCH= 16384;
constexpr int Lseq = 6;
constexpr int DFFc = 1024;
constexpr int DKh  = 32;
constexpr int KG   = 10;

// ---------- tiling ----------
constexpr int MB = 4;            // batches per block
constexpr int MT = MB * Lseq;    // 24 live rows
constexpr int MP = 32;           // padded rows (2 WMMA M-tiles)
constexpr int SA = 264;          // LDS stride (multiple of 8 bf16 -> 16B aligned rows)
constexpr int SQ = 264;
constexpr int SU = 72;           // FFN chunk buffer stride
constexpr int FFC = 64;          // FFN chunk width (4 n-tiles)

constexpr int NA_W = NL * 4 * D * D;   // attn weight elems
constexpr int NF_W = NL * D * DFFc;    // each ffn weight elems

// ---------- bf16 <-> f32 : native converts (v_cvt_*bf16*, RNE) ----------
__device__ __forceinline__ float tof(bf16 x) { return (float)x; }
__device__ __forceinline__ bf16  tobf(float f) { return (bf16)f; }

__device__ __forceinline__ float wave_sum(float v) {
  #pragma unroll
  for (int off = 16; off > 0; off >>= 1) v += __shfl_xor(v, off, 32);
  return v;
}

// ---------- WMMA fragment loaders ----------
// A (LDS, row-major): lanes 0-15 = rows M, K {0..7,16..23}; lanes 16-31: K {8..15,24..31}.
__device__ __forceinline__ v16bf load_afrag(const bf16* __restrict__ A, int lda,
                                            int mt, int ks, int lane) {
  const bf16* pa = A + (mt * 16 + (lane & 15)) * lda + ((lane >> 4) * 8) + ks * 32;
  v8bf lo = *(const v8bf*)pa;
  v8bf hi = *(const v8bf*)(pa + 16);
  return CAT16(lo, hi);
}
// B (global, pre-transposed [nOut][kIn]): lane&15 = column N,
// low-half lanes K {0..7,16..23}, high-half K {8..15,24..31}.
__device__ __forceinline__ v16bf load_bfrag(const bf16* __restrict__ BT, size_t ldb,
                                            int nt, int ks, int lane) {
  const bf16* pb = BT + (size_t)(nt * 16 + (lane & 15)) * ldb + ((lane >> 4) * 8) + ks * 32;
  v8bf lo = *(const v8bf*)pb;
  v8bf hi = *(const v8bf*)(pb + 16);
  return CAT16(lo, hi);
}
__device__ __forceinline__ v8f wmma_bf16(v16bf a, v16bf b, v8f c) {
  return __builtin_amdgcn_wmma_f32_16x16x32_bf16(false, a, false, b, (short)0, c,
                                                 false, false);
}

// C-tile layout: VGPR r, lanes 0-15 -> (M = r, N = lane); lanes 16-31 -> (M = 8+r).
__device__ __forceinline__ void store_tile_bf16(v8f acc, bf16* dst, int ldd, int mt,
                                                int ntLocal, int lane,
                                                const float* __restrict__ bias,
                                                int maxrow, bool relu) {
  const int m0 = mt * 16 + ((lane >> 4) << 3);
  const int nc = ntLocal * 16 + (lane & 15);
  const float bv = bias[nc];
  #pragma unroll
  for (int r = 0; r < 8; ++r) {
    const int m = m0 + r;
    float v = acc[r] + bv;
    if (relu) v = v > 0.f ? v : 0.f;
    if (m < maxrow) dst[m * ldd + nc] = tobf(v);
  }
}
__device__ __forceinline__ void store_tile_f32(v8f acc, float* dst, int ldd, int mt,
                                               int nt, int lane,
                                               const float* __restrict__ bias,
                                               int maxrow) {
  const int m0 = mt * 16 + ((lane >> 4) << 3);
  const int nc = nt * 16 + (lane & 15);
  const float bv = bias[nc];
  #pragma unroll
  for (int r = 0; r < 8; ++r) {
    const int m = m0 + r;
    if (m < maxrow) dst[m * ldd + nc] = acc[r] + bv;
  }
}

// torch-style LayerNorm: unbiased std (ddof=1), eps added to std.
__device__ __forceinline__ void ln_store_bf16(const float res[3][8],
                                              const float* __restrict__ la,
                                              const float* __restrict__ lb,
                                              bf16* dst, int ldd, int w, int lane) {
  #pragma unroll
  for (int rr = 0; rr < 3; ++rr) {
    const int r = w * 3 + rr;
    float s = 0.f;
    #pragma unroll
    for (int j = 0; j < 8; ++j) s += res[rr][j];
    const float mean = wave_sum(s) * (1.0f / 256.0f);
    float v = 0.f;
    #pragma unroll
    for (int j = 0; j < 8; ++j) { const float d0 = res[rr][j] - mean; v += d0 * d0; }
    const float inv = 1.0f / (sqrtf(wave_sum(v) * (1.0f / 255.0f)) + 1e-6f);
    #pragma unroll
    for (int j = 0; j < 8; ++j) {
      const int c = lane + 32 * j;
      dst[r * ldd + c] = tobf(la[c] * (res[rr][j] - mean) * inv + lb[c]);
    }
  }
}
__device__ __forceinline__ void ln_store_f32(const float res[3][8],
                                             const float* __restrict__ la,
                                             const float* __restrict__ lb,
                                             float* dst, int ldd, int w, int lane) {
  #pragma unroll
  for (int rr = 0; rr < 3; ++rr) {
    const int r = w * 3 + rr;
    float s = 0.f;
    #pragma unroll
    for (int j = 0; j < 8; ++j) s += res[rr][j];
    const float mean = wave_sum(s) * (1.0f / 256.0f);
    float v = 0.f;
    #pragma unroll
    for (int j = 0; j < 8; ++j) { const float d0 = res[rr][j] - mean; v += d0 * d0; }
    const float inv = 1.0f / (sqrtf(wave_sum(v) * (1.0f / 255.0f)) + 1e-6f);
    #pragma unroll
    for (int j = 0; j < 8; ++j) {
      const int c = lane + 32 * j;
      dst[r * ldd + c] = la[c] * (res[rr][j] - mean) * inv + lb[c];
    }
  }
}

// ---------- kernel A: gaussian positional encoding rows {0,50,...,250} ----------
__global__ __launch_bounds__(256) void posenc_kernel(const float* __restrict__ pe,
                                                     const float* __restrict__ mu,
                                                     const float* __restrict__ sg,
                                                     float* __restrict__ outp) {
  const int t = blockIdx.x;          // 0..5
  const int c = threadIdx.x;         // 0..255
  const float pos = (float)(t * 50);
  float lp[KG];
  float mx = -1e30f;
  #pragma unroll
  for (int k = 0; k < KG; ++k) {
    const float a = pos - mu[k];
    const float s = sg[k];
    lp[k] = -(a * a) / (2.f * s * s) - logf(s);
    mx = fmaxf(mx, lp[k]);
  }
  float den = 0.f;
  #pragma unroll
  for (int k = 0; k < KG; ++k) { lp[k] = expf(lp[k] - mx); den += lp[k]; }
  const float rd = 1.f / den;
  float acc = 0.f;
  #pragma unroll
  for (int k = 0; k < KG; ++k) acc += lp[k] * rd * pe[k * D + c];
  outp[t * D + c] = acc;
}

// ---------- kernel B: weight convert fp32->bf16 + transpose to [nOut][kIn] ----------
__global__ __launch_bounds__(256) void wconv_kernel(const float* __restrict__ aw,
                                                    const float* __restrict__ w1,
                                                    const float* __restrict__ w2,
                                                    bf16* __restrict__ ta,
                                                    bf16* __restrict__ t1,
                                                    bf16* __restrict__ t2) {
  const int idx = blockIdx.x * 256 + threadIdx.x;
  if (idx < NA_W) {                       // attn: [g][k][n] -> [g][n][k], g = layer*4+mat
    const int g = idx >> 16, rem = idx & 65535, nc = rem >> 8, kc = rem & 255;
    ta[idx] = tobf(aw[(g << 16) + (kc << 8) + nc]);
  } else if (idx < NA_W + NF_W) {         // w1: [n][d][ff] -> [n][ff][d]
    const int j = idx - NA_W;
    const int nl = j >> 18, rem = j & 262143, ff = rem >> 8, dc = rem & 255;
    t1[j] = tobf(w1[(nl << 18) + (dc << 10) + ff]);
  } else {                                // w2: [n][ff][d] -> [n][d][ff]
    const int j = idx - NA_W - NF_W;
    const int nl = j >> 18, rem = j & 262143, dc = rem >> 10, ff = rem & 1023;
    t2[j] = tobf(w2[(nl << 18) + (ff << 8) + dc]);
  }
}

// ---------- main fused transformer kernel ----------
__global__ __launch_bounds__(256) void trans_fused(
    const float* __restrict__ g_x,  const float* __restrict__ g_tr,
    const float* __restrict__ g_tw, const float* __restrict__ g_tb,
    const float* __restrict__ g_ab, const float* __restrict__ g_b1,
    const float* __restrict__ g_b2, const float* __restrict__ g_la,
    const float* __restrict__ g_lb, const float* __restrict__ g_fa,
    const float* __restrict__ g_fb, const float* __restrict__ g_pos,
    const bf16* __restrict__ g_wta, const bf16* __restrict__ g_w1t,
    const bf16* __restrict__ g_w2t, float* __restrict__ g_out) {
  __shared__ __align__(16) bf16 smem[MP * SA + 3 * MT * SQ];
  __shared__ float s_tr[MT];
  bf16*  abuf = smem;                 // [32][264] A-matrix (LN out / attn out)
  bf16*  qbuf = smem + MP * SA;       // [24][264]
  bf16*  kbuf = qbuf + MT * SQ;       // [24][264]
  bf16*  vbuf = kbuf + MT * SQ;       // [24][264]
  float* cbuf = (float*)qbuf;         // [24][256] f32, aliases q+k (sequenced by barriers)
  bf16*  ubuf = vbuf;                 // [32][72]  FFN chunk, aliases v

  const int tid  = threadIdx.x;
  const int w    = tid >> 5;
  const int lane = tid & 31;
  const int b0   = blockIdx.x;
  const int rb   = b0 * MT;           // global token-row base

  if (tid < MT) s_tr[tid] = g_tr[rb + tid];

  // ---- stage 0: bulk-stage the [24][256] f32 input tile into LDS via TDM ----
  const float* xsrc;
#if HAVE_TDM
  if (w == 0) {
    const unsigned lds_addr = (unsigned)(unsigned long long)(void*)cbuf;
    const unsigned long long ga =
        (unsigned long long)(const void*)(g_x + (size_t)rb * D);
    // D# group0: count=1 | lds_addr | global_addr[56:0] | type=2
    u32x4 g0 = { 1u, lds_addr, (unsigned)(ga & 0xFFFFFFFFull),
                 (unsigned)((ga >> 32) & 0x01FFFFFFull) | (2u << 30) };
    // D# group1: data_size=4B; tensor_dim0=256; tensor_dim1=BATCH*L-rb;
    //            tile 256 x 24; tensor_dim0_stride=256
    const unsigned td1 = (unsigned)(BATCH * Lseq - rb);
    i32x8 g1 = { (int)(2u << 16),
                 (int)(256u << 16),
                 (int)((256u >> 16) | ((td1 & 0xFFFFu) << 16)),
                 (int)((td1 >> 16) | (256u << 16)),
                 (int)MT,
                 (int)256,
                 0, 0 };
    i32x4 z4 = { 0, 0, 0, 0 };
#if __clang_major__ >= 23
    i32x8 z8 = { 0, 0, 0, 0, 0, 0, 0, 0 };
    __builtin_amdgcn_tensor_load_to_lds(g0, g1, z4, z4, z8, 0);
#else
    __builtin_amdgcn_tensor_load_to_lds(g0, g1, z4, z4, 0);
#endif
    __builtin_amdgcn_s_wait_tensorcnt(0);
  }
  xsrc = cbuf;
#else
  xsrc = g_x + (size_t)rb * D;
#endif
  __syncthreads();

  // ---- x = staged tile + pos_enc + cos(time*w + b); residual lives in registers ----
  float res[3][8];
  #pragma unroll
  for (int rr = 0; rr < 3; ++rr) {
    const int r = w * 3 + rr;
    const int t = r % Lseq;
    const float tm = g_tr[rb + r];
    #pragma unroll
    for (int j = 0; j < 8; ++j) {
      const int c = lane + 32 * j;
      res[rr][j] = xsrc[r * D + c] + g_pos[t * D + c] +
                   cosf(tm * g_tw[c] + g_tb[c]);
    }
  }
  __syncthreads();   // everyone done reading cbuf before QKV overwrites it

  const v8f vzero = {0.f, 0.f, 0.f, 0.f, 0.f, 0.f, 0.f, 0.f};

  for (int n = 0; n < NL; ++n) {
    const bf16*  wta = g_wta + (size_t)n * 4 * D * D;
    const bf16*  w1t = g_w1t + (size_t)n * D * DFFc;
    const bf16*  w2t = g_w2t + (size_t)n * D * DFFc;
    const float* ab  = g_ab + (size_t)n * 4 * D;

    // ---- LN1 -> abuf ----
    ln_store_bf16(res, g_la + (n * 2 + 0) * D, g_lb + (n * 2 + 0) * D, abuf, SA, w, lane);
    __syncthreads();

    // ---- Q,K,V projections: 48 (which,nt) panels over 8 waves; reuse B over 2 M-tiles
    for (int p = w; p < 48; p += 8) {
      const int which = p >> 4, nt = p & 15;
      const bf16* BT = wta + (size_t)which * D * D;
      v16bf bfr[8];
      #pragma unroll
      for (int ks = 0; ks < 8; ++ks) bfr[ks] = load_bfrag(BT, D, nt, ks, lane);
      bf16* dst = (which == 0) ? qbuf : (which == 1) ? kbuf : vbuf;
      #pragma unroll
      for (int mt = 0; mt < 2; ++mt) {
        v8f acc = vzero;
        #pragma unroll
        for (int ks = 0; ks < 8; ++ks)
          acc = wmma_bf16(load_afrag(abuf, SA, mt, ks, lane), bfr[ks], acc);
        store_tile_bf16(acc, dst, SQ, mt, nt, lane, ab + which * D, MT, false);
      }
    }
    __syncthreads();

    // ---- attention (per batch/head/query, L=6) -> abuf ----
    if (tid < MB * Hh * Lseq) {       // 192 threads
      const int bi = tid / (Hh * Lseq);
      const int hp = (tid / Lseq) % Hh;
      const int tq = tid % Lseq;
      const int rq = bi * Lseq + tq;
      float q[DKh];
      #pragma unroll
      for (int d0 = 0; d0 < DKh; ++d0) q[d0] = tof(qbuf[rq * SQ + hp * DKh + d0]);
      float sc[Lseq];
      const float tbq = s_tr[rq];
      #pragma unroll
      for (int tk = 0; tk < Lseq; ++tk) {
        const int rk = bi * Lseq + tk;
        if (tbq != 0.f && s_tr[rk] != 0.f && tk <= tq) {
          float s = 0.f;
          #pragma unroll
          for (int d0 = 0; d0 < DKh; ++d0) s += q[d0] * tof(kbuf[rk * SQ + hp * DKh + d0]);
          sc[tk] = s * 0.17677669529663687f;   // 1/sqrt(32)
        } else {
          sc[tk] = -1e9f;
        }
      }
      float mx = sc[0];
      #pragma unroll
      for (int tk = 1; tk < Lseq; ++tk) mx = fmaxf(mx, sc[tk]);
      float den = 0.f;
      #pragma unroll
      for (int tk = 0; tk < Lseq; ++tk) { sc[tk] = expf(sc[tk] - mx); den += sc[tk]; }
      const float rden = 1.f / den;
      float o[DKh];
      #pragma unroll
      for (int d0 = 0; d0 < DKh; ++d0) o[d0] = 0.f;
      #pragma unroll
      for (int tk = 0; tk < Lseq; ++tk) {
        const float p = sc[tk] * rden;
        const bf16* vp = vbuf + (bi * Lseq + tk) * SQ + hp * DKh;
        #pragma unroll
        for (int d0 = 0; d0 < DKh; ++d0) o[d0] += p * tof(vp[d0]);
      }
      #pragma unroll
      for (int d0 = 0; d0 < DKh; ++d0)
        abuf[rq * SA + hp * DKh + d0] = tobf(o[d0]);
    }
    __syncthreads();

    // ---- O projection -> cbuf (B reused over both M-tiles), then residual ----
    for (int p = w; p < 16; p += 8) {
      const bf16* BT = wta + (size_t)3 * D * D;
      v16bf bfr[8];
      #pragma unroll
      for (int ks = 0; ks < 8; ++ks) bfr[ks] = load_bfrag(BT, D, p, ks, lane);
      #pragma unroll
      for (int mt = 0; mt < 2; ++mt) {
        v8f acc = vzero;
        #pragma unroll
        for (int ks = 0; ks < 8; ++ks)
          acc = wmma_bf16(load_afrag(abuf, SA, mt, ks, lane), bfr[ks], acc);
        store_tile_f32(acc, cbuf, D, mt, p, lane, ab + 3 * D, MT);
      }
    }
    __syncthreads();
    #pragma unroll
    for (int rr = 0; rr < 3; ++rr)
      #pragma unroll
      for (int j = 0; j < 8; ++j)
        res[rr][j] += cbuf[(w * 3 + rr) * D + lane + 32 * j];

    // ---- LN2 -> abuf ----
    ln_store_bf16(res, g_la + (n * 2 + 1) * D, g_lb + (n * 2 + 1) * D, abuf, SA, w, lane);
    __syncthreads();

    // ---- FFN, chunked over DFF (64 cols/chunk), acc2 persistent in regs ----
    v8f acc2[4] = {vzero, vzero, vzero, vzero};
    for (int c = 0; c < DFFc / FFC; ++c) {
      {
        const int mt = w >> 2, ntc = w & 3;
        v8f acc = vzero;
        #pragma unroll
        for (int ks = 0; ks < 8; ++ks)
          acc = wmma_bf16(load_afrag(abuf, SA, mt, ks, lane),
                          load_bfrag(w1t, D, c * 4 + ntc, ks, lane), acc);
        store_tile_bf16(acc, ubuf, SU, mt, ntc, lane, g_b1 + n * DFFc + c * FFC, MT, true);
      }
      if (c + 1 < DFFc / FFC) {  // pull next weight panels toward the caches
        __builtin_prefetch(w1t + (size_t)(c + 1) * FFC * D + (size_t)lane * 2 * D, 0, 0);
        __builtin_prefetch(w2t + (size_t)(c + 1) * FFC + (size_t)lane * 8 * DFFc, 0, 0);
      }
      __syncthreads();
      #pragma unroll
      for (int i = 0; i < 2; ++i) {           // nt = w, w+8: reuse B over both M-tiles
        const int nt = w + 8 * i;
        v16bf bf0 = load_bfrag(w2t + c * FFC, DFFc, nt, 0, lane);
        v16bf bf1 = load_bfrag(w2t + c * FFC, DFFc, nt, 1, lane);
        #pragma unroll
        for (int mt = 0; mt < 2; ++mt) {
          v8f a = acc2[i * 2 + mt];
          a = wmma_bf16(load_afrag(ubuf, SU, mt, 0, lane), bf0, a);
          a = wmma_bf16(load_afrag(ubuf, SU, mt, 1, lane), bf1, a);
          acc2[i * 2 + mt] = a;
        }
      }
      __syncthreads();
    }
    #pragma unroll
    for (int i = 0; i < 2; ++i)
      #pragma unroll
      for (int mt = 0; mt < 2; ++mt)
        store_tile_f32(acc2[i * 2 + mt], cbuf, D, mt, w + 8 * i, lane, g_b2 + n * D, MT);
    __syncthreads();
    #pragma unroll
    for (int rr = 0; rr < 3; ++rr)
      #pragma unroll
      for (int j = 0; j < 8; ++j)
        res[rr][j] += cbuf[(w * 3 + rr) * D + lane + 32 * j];
  }

  // ---- final LN -> cbuf, then mean over L per batch ----
  ln_store_f32(res, g_fa, g_fb, cbuf, D, w, lane);
  __syncthreads();
  #pragma unroll
  for (int e = 0; e < 4; ++e) {
    const int qi = e * 256 + tid;
    const int bb = qi >> 8, c = qi & 255;
    float s = 0.f;
    #pragma unroll
    for (int t = 0; t < Lseq; ++t) s += cbuf[(bb * Lseq + t) * D + c];
    g_out[(size_t)(b0 * MB + bb) * D + c] = s * (1.0f / 6.0f);
  }
}

// ---------- launch ----------
extern "C" void kernel_launch(void* const* d_in, const int* in_sizes, int n_in,
                              void* d_out, int out_size, void* d_ws, size_t ws_size,
                              hipStream_t stream) {
  const float* trans = (const float*)d_in[0];
  const float* trec  = (const float*)d_in[1];
  // d_in[2] = seed (identity dropout in eval) -- unused
  const float* pe    = (const float*)d_in[3];
  const float* mu    = (const float*)d_in[4];
  const float* sg    = (const float*)d_in[5];
  const float* tw    = (const float*)d_in[6];
  const float* tb    = (const float*)d_in[7];
  const float* aw    = (const float*)d_in[8];
  const float* ab    = (const float*)d_in[9];
  const float* w1    = (const float*)d_in[10];
  const float* b1    = (const float*)d_in[11];
  const float* w2    = (const float*)d_in[12];
  const float* b2    = (const float*)d_in[13];
  const float* la    = (const float*)d_in[14];
  const float* lb    = (const float*)d_in[15];
  const float* fa    = (const float*)d_in[16];
  const float* fb    = (const float*)d_in[17];
  float* outp = (float*)d_out;

  // workspace layout: [posenc f32 | attnW^T bf16 | w1^T bf16 | w2^T bf16]
  float* wspos = (float*)d_ws;
  bf16* ta = (bf16*)((char*)d_ws + 16384);
  bf16* t1 = (bf16*)((char*)d_ws + 16384 + (size_t)NA_W * 2);
  bf16* t2 = (bf16*)((char*)d_ws + 16384 + (size_t)NA_W * 2 + (size_t)NF_W * 2);

  posenc_kernel<<<Lseq, 256, 0, stream>>>(pe, mu, sg, wspos);
  wconv_kernel<<<(NA_W + 2 * NF_W) / 256, 256, 0, stream>>>(aw, w1, w2, ta, t1, t2);
  trans_fused<<<BATCH / MB, 256, 0, stream>>>(trans, trec, tw, tb, ab, b1, b2,
                                              la, lb, fa, fb, wspos, ta, t1, t2, outp);
}
